// RgbtDeformableTransformerAttention_40303973106032
// MI455X (gfx1250) — matrix operand order repaired and checked by reading, not compiled, not run on hardware
//
#include <hip/hip_runtime.h>
#include <hip/hip_bf16.h>

// ---------------- problem constants (from reference) ----------------
#define BB   4
#define DD   256
#define DFF  1024
#define NH   8
#define NP   4
#define HD   32           // DD / NH
#define H0   96
#define W0   96
#define LEN  (H0 * W0)    // 9216
#define MM   (BB * LEN)   // 36864 rows for every GEMM

typedef unsigned short u16;
typedef __bf16 bf16;
typedef __attribute__((ext_vector_type(16))) __bf16  v16bf;
typedef __attribute__((ext_vector_type(8)))  float   v8f;
typedef __attribute__((ext_vector_type(4)))  unsigned int u32x4;

union ABFrag { u32x4 u[2]; v16bf v; };

// round-to-nearest-even f32 -> bf16 (as raw u16)
__device__ __forceinline__ u16 f2bf(float f) {
    unsigned u = __float_as_uint(f);
    unsigned r = ((u >> 16) & 1u) + 0x7FFFu;
    return (u16)((u + r) >> 16);
}

// ---------------- f32 -> bf16 conversion ----------------
__global__ __launch_bounds__(256) void k_cvt_bf16(const float* __restrict__ in,
                                                  u16* __restrict__ out, int n) {
    int i = blockIdx.x * 256 + threadIdx.x;
    if (i < n) out[i] = f2bf(in[i]);
}

// ---------------- WMMA bf16 GEMM: C = A(MxK) * W(NxK)^T + bias ----------------
// Block = 8 waves arranged WM x WN. A strip (WM*16 rows) is staged into LDS in
// 64-wide K chunks with double-buffered async global->LDS loads (ASYNCcnt);
// each wave computes a 16 x (16*NT) tile via v_wmma_f32_16x16x32_bf16, reading
// A fragments from LDS (ds_load_b128) and B fragments direct from global
// (weights are L2-resident).
template<int NT, int WN, bool RELU, bool OUTBF16>
__global__ __launch_bounds__(256) void k_gemm_bf16(
    const u16* __restrict__ A, const u16* __restrict__ W,
    const float* __restrict__ bias,
    float* __restrict__ Cf, u16* __restrict__ Cb,
    int M, int N, int K)
{
    constexpr int WM   = 8 / WN;
    constexpr int ROWS = WM * 16;
    constexpr int CK   = 64;                       // K-chunk (bf16 elements)
    __shared__ u16 tile[2][ROWS][CK];

    const int lane = threadIdx.x & 31;
    const int wave = threadIdx.x >> 5;
    const int wn   = wave % WN;
    const int wm   = wave / WN;

    const int bn   = N / (WN * NT * 16);           // blocks along N
    const int mblk = (blockIdx.x / bn) * ROWS;
    const int n0   = ((blockIdx.x % bn) * WN + wn) * NT * 16;
    const int lm   = lane & 15;
    const int hi   = lane >> 4;                    // half-wave (16-bit frag layout)

    const unsigned long long Abase = (unsigned long long)(size_t)A;
    const int srow = threadIdx.x >> 3;             // 32 rows covered per pass
    const int sseg = threadIdx.x & 7;              // 8 x 16B segments per row

    // async global->LDS staging of one K-chunk of the A strip
    auto stage = [&](int buf, int kc) {
        for (int r = srow; r < ROWS; r += 32) {
            unsigned lds  = (unsigned)(size_t)&tile[buf][r][sseg * 8];
            unsigned voff = (unsigned)(((size_t)(mblk + r) * K + kc + sseg * 8) * 2);
            asm volatile("global_load_async_to_lds_b128 %0, %1, %2"
                         :: "v"(lds), "v"(voff), "s"(Abase) : "memory");
        }
    };

    v8f acc[NT];
#pragma unroll
    for (int t = 0; t < NT; ++t) acc[t] = v8f{0.f,0.f,0.f,0.f,0.f,0.f,0.f,0.f};

    const int nchunk = K / CK;
    stage(0, 0);
    asm volatile("s_wait_asynccnt 0x0" ::: "memory");
    __syncthreads();

    for (int c = 0; c < nchunk; ++c) {
        if (c + 1 < nchunk) stage((c + 1) & 1, (c + 1) * CK);   // prefetch next chunk
        const u16* arow = &tile[c & 1][wm * 16 + lm][0];
#pragma unroll
        for (int kk = 0; kk < CK; kk += 32) {
            // A fragment from LDS: lane<16 -> K {kk..+15}; lane>=16 -> {+16..+31}
            ABFrag af;
            af.u[0] = *(const u32x4*)(arow + kk + hi * 16);
            af.u[1] = *(const u32x4*)(arow + kk + hi * 16 + 8);
            const int kg = c * CK + kk;
#pragma unroll
            for (int t = 0; t < NT; ++t) {
                // B fragment: lane = column, K contiguous (lanes 0-15: K kg..+15, 16-31: +16..+31)
                const u16* brow = W + (size_t)(n0 + t * 16 + lm) * K + kg + hi * 16;
                ABFrag bfv;
                bfv.u[0] = *(const u32x4*)(brow);
                bfv.u[1] = *(const u32x4*)(brow + 8);
                acc[t] = __builtin_amdgcn_wmma_f32_16x16x32_bf16(
                    false, af.v, false, bfv.v, (short)0, acc[t], false, false);
            }
        }
        asm volatile("s_wait_asynccnt 0x0" ::: "memory");       // chunk c+1 landed
        __syncthreads();                                        // tile consumed by all waves
    }

    // C/D layout: lane L, vgpr i -> row (L>>4)*8 + i, col (L&15)
    const int m0 = mblk + wm * 16;
#pragma unroll
    for (int t = 0; t < NT; ++t) {
        const int col = n0 + t * 16 + lm;
        const float bs = bias[col];
#pragma unroll
        for (int i = 0; i < 8; ++i) {
            float v = acc[t][i] + bs;
            if (RELU) v = v > 0.f ? v : 0.f;
            size_t off = (size_t)(m0 + hi * 8 + i) * N + col;
            if (OUTBF16) Cb[off] = f2bf(v);
            else         Cf[off] = v;
        }
    }
}

// ---------------- softmax over NP=4 points per (b*q, head) ----------------
__global__ __launch_bounds__(256) void k_softmax4(float* __restrict__ attn) {
    int i = blockIdx.x * 256 + threadIdx.x;           // i = bq*NH + h
    if (i >= MM * NH) return;
    float* p = attn + (size_t)(i >> 3) * (NH * NP) + (i & 7) * NP;
    float a0 = p[0], a1 = p[1], a2 = p[2], a3 = p[3];
    float m = fmaxf(fmaxf(a0, a1), fmaxf(a2, a3));
    float e0 = __expf(a0 - m), e1 = __expf(a1 - m), e2 = __expf(a2 - m), e3 = __expf(a3 - m);
    float inv = 1.f / (e0 + e1 + e2 + e3);
    p[0] = e0 * inv; p[1] = e1 * inv; p[2] = e2 * inv; p[3] = e3 * inv;
}

// ---------------- deformable bilinear sampling ----------------
// one wave per (b, q, h); lane = channel d in [0,32)
__global__ __launch_bounds__(256) void k_sample(
    const float* __restrict__ value,   // (B*LEN, 256) = (b,pix,h,d)
    const float* __restrict__ off,     // (B*LEN, 64)  = (h,p,2)
    const float* __restrict__ attn,    // (B*LEN, 32)  probs, (h,p)
    u16* __restrict__ samp)            // (B*LEN, 256) bf16
{
    const int lane = threadIdx.x & 31;
    const int item = blockIdx.x * 8 + (threadIdx.x >> 5);   // [0, MM*NH)
    const int h  = item & 7;
    const int bq = item >> 3;
    const int q  = bq % LEN;
    const int b  = bq / LEN;
    const int px = q % W0, py = q / W0;

    const float* vb   = value + (size_t)b * LEN * DD + h * HD + lane;
    const float* orow = off  + (size_t)bq * (NH * NP * 2) + h * (NP * 2);
    const float* arow = attn + (size_t)bq * (NH * NP)     + h * NP;

    float acc = 0.f;
#pragma unroll
    for (int p = 0; p < NP; ++p) {
        // ref point collapses: x = (rx + ox/W)*W - 0.5 = px + ox  (96x96 level)
        float x = (float)px + orow[p * 2 + 0];
        float y = (float)py + orow[p * 2 + 1];
        float a = arow[p];
        float x0f = floorf(x), y0f = floorf(y);
        int   x0 = (int)x0f,  y0 = (int)y0f;
        float wx1 = x - x0f, wy1 = y - y0f;
        float wx0 = 1.f - wx1, wy0 = 1.f - wy1;

        float s = 0.f;
#pragma unroll
        for (int cy = 0; cy < 2; ++cy) {
#pragma unroll
            for (int cx = 0; cx < 2; ++cx) {
                int xi = x0 + cx, yi = y0 + cy;
                float w = (cx ? wx1 : wx0) * (cy ? wy1 : wy0);
                bool valid = (xi >= 0) & (xi < W0) & (yi >= 0) & (yi < H0);
                int xc = xi < 0 ? 0 : (xi > W0 - 1 ? W0 - 1 : xi);
                int yc = yi < 0 ? 0 : (yi > H0 - 1 ? H0 - 1 : yi);
                float g = vb[(size_t)(yc * W0 + xc) * DD];
                s += valid ? g * w : 0.f;
            }
        }
        acc += a * s;
    }
    samp[(size_t)bq * DD + h * HD + lane] = f2bf(acc);
}

// ---------------- residual add + LayerNorm over D=256 ----------------
__global__ __launch_bounds__(256) void k_add_ln(
    const float* __restrict__ x, const float* __restrict__ resid,
    const float* __restrict__ g, const float* __restrict__ be,
    float* __restrict__ outf, u16* __restrict__ outb /* may be null */)
{
    __shared__ float s_sum[8], s_sq[8];
    const int row = blockIdx.x;
    const int t   = threadIdx.x;
    const size_t base = (size_t)row * DD;
    float v = x[base + t] + resid[base + t];
    float s = v, sq = v * v;
#pragma unroll
    for (int o = 16; o > 0; o >>= 1) {
        s  += __shfl_down(s,  o, 32);
        sq += __shfl_down(sq, o, 32);
    }
    if ((t & 31) == 0) { s_sum[t >> 5] = s; s_sq[t >> 5] = sq; }
    __syncthreads();
    float tot = 0.f, totq = 0.f;
#pragma unroll
    for (int i = 0; i < 8; ++i) { tot += s_sum[i]; totq += s_sq[i]; }
    float mu  = tot * (1.f / DD);
    float var = totq * (1.f / DD) - mu * mu;
    float r   = rsqrtf(var + 1e-5f);
    float y   = (v - mu) * r * g[t] + be[t];
    outf[base + t] = y;
    if (outb) outb[base + t] = f2bf(y);
}

// ---------------- host-side pipeline ----------------
extern "C" void kernel_launch(void* const* d_in, const int* in_sizes, int n_in,
                              void* d_out, int out_size, void* d_ws, size_t ws_size,
                              hipStream_t stream) {
    const float* reference = (const float*)d_in[0];
    const float* sensed    = (const float*)d_in[1];
    // d_in[2] = spatial_shapes (compile-time constants here)
    const float* W_off  = (const float*)d_in[3];
    const float* b_off  = (const float*)d_in[4];
    const float* W_attn = (const float*)d_in[5];
    const float* b_attn = (const float*)d_in[6];
    const float* W_val  = (const float*)d_in[7];
    const float* b_val  = (const float*)d_in[8];
    const float* W_out  = (const float*)d_in[9];
    const float* b_out  = (const float*)d_in[10];
    const float* g1     = (const float*)d_in[11];
    const float* be1    = (const float*)d_in[12];
    const float* W1     = (const float*)d_in[13];
    const float* b1     = (const float*)d_in[14];
    const float* W2     = (const float*)d_in[15];
    const float* b2     = (const float*)d_in[16];
    const float* g2     = (const float*)d_in[17];
    const float* be2    = (const float*)d_in[18];

    // ---- carve workspace ----
    char* p = (char*)d_ws;
    auto carve = [&p](size_t bytes) { char* r = p; p += (bytes + 255) & ~(size_t)255; return r; };
    u16*   q_bf     = (u16*)  carve((size_t)MM * DD * 2);
    u16*   s_bf     = (u16*)  carve((size_t)MM * DD * 2);
    u16*   Wval_bf  = (u16*)  carve((size_t)DD * DD * 2);
    u16*   Woff_bf  = (u16*)  carve((size_t)64 * DD * 2);
    u16*   Wattn_bf = (u16*)  carve((size_t)32 * DD * 2);
    u16*   Wout_bf  = (u16*)  carve((size_t)DD * DD * 2);
    u16*   W1_bf    = (u16*)  carve((size_t)DFF * DD * 2);
    u16*   W2_bf    = (u16*)  carve((size_t)DD * DFF * 2);
    float* tgt_f    = (float*)carve((size_t)MM * DD * 4);
    u16*   tgt_bf   = (u16*)  carve((size_t)MM * DD * 2);
    float* ffn_f    = (float*)carve((size_t)MM * DD * 4);
    float* aout_f   = (float*)carve((size_t)MM * DD * 4);
    // region reused: {value, off, attn, samp} live only until out-proj; then FFN hidden
    char*  region   = carve((size_t)MM * DFF * 2);       // 75.5 MB
    float* value_f  = (float*)region;                                   // MM*256 f32
    float* off_f    = (float*)(region + (size_t)MM * DD * 4);           // MM*64 f32
    float* attn_f   = (float*)(region + (size_t)MM * DD * 4 + (size_t)MM * 64 * 4);
    u16*   samp_bf  = (u16*)  (region + (size_t)MM * DD * 4 + (size_t)MM * 64 * 4 + (size_t)MM * 32 * 4);
    u16*   hid_bf   = (u16*)region;                                     // MM*DFF bf16

    auto cvt = [&](const float* src, u16* dst, int n) {
        k_cvt_bf16<<<(n + 255) / 256, 256, 0, stream>>>(src, dst, n);
    };
    // ---- stage 0: bf16 conversions ----
    cvt(reference, q_bf, MM * DD);
    cvt(sensed,    s_bf, MM * DD);
    cvt(W_val,  Wval_bf,  DD * DD);
    cvt(W_off,  Woff_bf,  64 * DD);
    cvt(W_attn, Wattn_bf, 32 * DD);
    cvt(W_out,  Wout_bf,  DD * DD);
    cvt(W1,     W1_bf,    DFF * DD);
    cvt(W2,     W2_bf,    DD * DFF);

    // grid = (M / (WM*16)) * (N / (WN*NT*16)),  WM = 8/WN
    auto gblk = [](int M, int N, int NT, int WN) {
        int WM = 8 / WN;
        return (M / (WM * 16)) * (N / (WN * NT * 16));
    };

    // ---- value projection: (MM,256) = sensed @ W_val^T + b_val ----
    k_gemm_bf16<4, 4, false, false><<<gblk(MM, DD, 4, 4), 256, 0, stream>>>(
        s_bf, Wval_bf, b_val, value_f, nullptr, MM, DD, DD);
    // ---- sampling offsets: (MM,64) ----
    k_gemm_bf16<1, 4, false, false><<<gblk(MM, 64, 1, 4), 256, 0, stream>>>(
        q_bf, Woff_bf, b_off, off_f, nullptr, MM, 64, DD);
    // ---- attention logits: (MM,32) ----
    k_gemm_bf16<1, 2, false, false><<<gblk(MM, 32, 1, 2), 256, 0, stream>>>(
        q_bf, Wattn_bf, b_attn, attn_f, nullptr, MM, 32, DD);
    // ---- softmax over 4 points ----
    k_softmax4<<<(MM * NH + 255) / 256, 256, 0, stream>>>(attn_f);
    // ---- deformable bilinear sampling -> bf16 samp ----
    k_sample<<<(MM * NH) / 8, 256, 0, stream>>>(value_f, off_f, attn_f, samp_bf);
    // ---- output projection: (MM,256) = samp @ W_out^T + b_out ----
    k_gemm_bf16<4, 4, false, false><<<gblk(MM, DD, 4, 4), 256, 0, stream>>>(
        samp_bf, Wout_bf, b_out, aout_f, nullptr, MM, DD, DD);
    // ---- residual + LayerNorm1 -> tgt (f32 + bf16) ----
    k_add_ln<<<MM, 256, 0, stream>>>(aout_f, reference, g1, be1, tgt_f, tgt_bf);
    // ---- FFN1: (MM,1024) = relu(tgt @ W1^T + b1), bf16 out (overwrites dead region) ----
    k_gemm_bf16<4, 4, true, true><<<gblk(MM, DFF, 4, 4), 256, 0, stream>>>(
        tgt_bf, W1_bf, b1, nullptr, hid_bf, MM, DFF, DD);
    // ---- FFN2: (MM,256) = hid @ W2^T + b2 ----
    k_gemm_bf16<4, 4, false, false><<<gblk(MM, DD, 4, 4), 256, 0, stream>>>(
        hid_bf, W2_bf, b2, ffn_f, nullptr, MM, DD, DFF);
    // ---- residual + LayerNorm2 -> final output ----
    k_add_ln<<<MM, 256, 0, stream>>>(ffn_f, tgt_f, g2, be2, (float*)d_out, nullptr);
}
